// GCN2_30210799960820
// MI455X (gfx1250) — matrix operand order repaired and checked by reading, not compiled, not run on hardware
//
#include <hip/hip_runtime.h>

typedef __attribute__((ext_vector_type(16))) _Float16 v16h;
typedef __attribute__((ext_vector_type(8)))  float    v8f;

#define F_IN 128
#define H1D  64
#define H2D  64
#define CD   16

// ---------------------------------------------------------------------------
// Pack W (row-major f32, D_IN x D_OUT) into WMMA B-fragment order as f16.
// Layout: WB[((ks*CT + c)*32 + lane)*16 + j] = W[ks*32 + (j<8 ? khalf*8 + j : 16 + khalf*8 + j-8)][c*16 + lrow]
// so a wave's B fragment is one contiguous 32B v16h load per lane.
// ---------------------------------------------------------------------------
template<int D_IN, int D_OUT>
__global__ __launch_bounds__(256)
void pack_w(const float* __restrict__ W, _Float16* __restrict__ WB)
{
    constexpr int CT = D_OUT / 16;
    const int gid = blockIdx.x * blockDim.x + threadIdx.x;
    if (gid >= D_IN * D_OUT) return;
    const int j    = gid & 15;
    const int lane = (gid >> 4) & 31;
    const int idx  = gid >> 9;          // fragment index = ks*CT + c
    const int c    = idx % CT;
    const int ks   = idx / CT;
    const int khalf = lane >> 4;
    const int lrow  = lane & 15;
    const int k = ks * 32 + (j < 8 ? khalf * 8 + j : 16 + khalf * 8 + (j - 8));
    const int n = c * 16 + lrow;
    WB[gid] = (_Float16)W[(size_t)k * D_OUT + n];
}

// ---------------------------------------------------------------------------
// Dense GEMM: out[n_rows x D_OUT] = (RELU ? relu(A) : A)[n_rows x D_IN] @ W
// One wave per 16-row stripe; A fragment loaded once per K-step (4x b128),
// reused across all D_OUT/16 column tiles; B fragments are pre-swizzled f16
// (single v16h load each). Fragment layouts per CDNA5 ISA 7.12.2 (wave32).
// ---------------------------------------------------------------------------
template<int D_IN, int D_OUT, bool RELU>
__global__ __launch_bounds__(256)
void gemm_wmma(const float* __restrict__ A, const _Float16* __restrict__ WB,
               float* __restrict__ out, int n_rows)
{
    constexpr int CT = D_OUT / 16;                 // column tiles per wave
    const int lane  = threadIdx.x & 31;
    const int wave  = blockIdx.x * (blockDim.x >> 5) + (threadIdx.x >> 5);
    const int row_tile = wave;
    if (row_tile * 16 >= n_rows) return;           // wave-uniform guard (EXEC stays full)

    const int khalf = lane >> 4;                   // 0: low K half, 1: high K half
    const int lrow  = lane & 15;

    v8f acc[CT];
    #pragma unroll
    for (int c = 0; c < CT; ++c)
        #pragma unroll
        for (int r = 0; r < 8; ++r) acc[c][r] = 0.0f;

    const float* arow = A + (size_t)(row_tile * 16 + lrow) * D_IN;

    #pragma unroll
    for (int ks = 0; ks < D_IN / 32; ++ks) {
        const int k0 = ks * 32;
        // ---- A fragment: 4x float4 (b128), then convert ----
        float a0[16];
        const float4* apLo = (const float4*)(arow + k0 + khalf * 8);
        const float4* apHi = (const float4*)(arow + k0 + 16 + khalf * 8);
        *(float4*)(a0 + 0)  = apLo[0];
        *(float4*)(a0 + 4)  = apLo[1];
        *(float4*)(a0 + 8)  = apHi[0];
        *(float4*)(a0 + 12) = apHi[1];
        v16h afrag;
        #pragma unroll
        for (int j = 0; j < 16; ++j) {
            float x = a0[j];
            if (RELU) x = fmaxf(x, 0.0f);
            afrag[j] = (_Float16)x;
        }
        // ---- one WMMA per column tile; B = single contiguous v16h ----
        #pragma unroll
        for (int c = 0; c < CT; ++c) {
            const v16h bfrag = *(const v16h*)(WB + (((size_t)(ks * CT + c) << 5) + lane) * 16);
            acc[c] = __builtin_amdgcn_wmma_f32_16x16x32_f16(
                false, afrag, false, bfrag, (short)0, acc[c], false, false);
        }
    }

    #pragma unroll
    for (int c = 0; c < CT; ++c) {
        const int n = c * 16 + lrow;
        #pragma unroll
        for (int r = 0; r < 8; ++r) {
            const int row = row_tile * 16 + khalf * 8 + r;
            out[(size_t)row * D_OUT + n] = acc[c][r];
        }
    }
}

// ---------------------------------------------------------------------------
// agg[:] = broadcast(bias)  (also fully overwrites poisoned buffers)
// ---------------------------------------------------------------------------
template<int D>
__global__ __launch_bounds__(256)
void init_bias(float* __restrict__ out, const float* __restrict__ b, long long total)
{
    long long gid = (long long)blockIdx.x * blockDim.x + threadIdx.x;
    if (gid >= total) return;
    out[gid] = b[(int)(gid & (D - 1))];
}

// ---------------------------------------------------------------------------
// Edge-parallel SpMM scatter: agg[dst[e]] += support[src[e]] * val[e]
// D/4 threads per edge, float4 gather (L2-resident), f32 atomic adds.
// ---------------------------------------------------------------------------
template<int D>
__global__ __launch_bounds__(256)
void spmm_scatter(const float* __restrict__ support, const float* __restrict__ val,
                  const int* __restrict__ src, const int* __restrict__ dst,
                  float* __restrict__ agg, long long total)
{
    constexpr int TPE = D / 4;
    long long gid = (long long)blockIdx.x * blockDim.x + threadIdx.x;
    if (gid >= total) return;
    const int e = (int)(gid / TPE);
    const int f = (int)(gid % TPE) * 4;
    const int s = src[e];
    const int d = dst[e];
    const float v = val[e];
    const float4 xv = *(const float4*)(support + (size_t)s * D + f);
    float* p = agg + (size_t)d * D + f;
    atomicAdd(p + 0, xv.x * v);
    atomicAdd(p + 1, xv.y * v);
    atomicAdd(p + 2, xv.z * v);
    atomicAdd(p + 3, xv.w * v);
}

// ---------------------------------------------------------------------------
// inputs: 0=x 1=edge_val 2=W1 3=b1 4=W2 5=b2 6=W3 7=b3 8=edge_src 9=edge_dst
// ---------------------------------------------------------------------------
extern "C" void kernel_launch(void* const* d_in, const int* in_sizes, int n_in,
                              void* d_out, int out_size, void* d_ws, size_t ws_size,
                              hipStream_t stream) {
    const float* x    = (const float*)d_in[0];
    const float* ev   = (const float*)d_in[1];
    const float* W1   = (const float*)d_in[2];
    const float* b1   = (const float*)d_in[3];
    const float* W2   = (const float*)d_in[4];
    const float* b2   = (const float*)d_in[5];
    const float* W3   = (const float*)d_in[6];
    const float* b3   = (const float*)d_in[7];
    const int*   esrc = (const int*)d_in[8];
    const int*   edst = (const int*)d_in[9];
    float*       out  = (float*)d_out;

    const int n = in_sizes[0] / F_IN;   // 100000 (multiple of 16)
    const int e = in_sizes[8];          // 1000000

    float* bufA = (float*)d_ws;                  // support buffer, n x 64
    float* bufB = bufA + (size_t)n * H1D;        // agg buffer,     n x 64
    float* bufC = bufB + (size_t)n * H1D;        // support3,       n x 16
    _Float16* wb1 = (_Float16*)(bufC + (size_t)n * CD);   // packed W1 (f16)
    _Float16* wb2 = wb1 + F_IN * H1D;                     // packed W2
    _Float16* wb3 = wb2 + H1D * H2D;                      // packed W3

    const dim3 B(256);
    const int row_tiles   = (n + 15) / 16;
    const int gemm_blocks = (row_tiles + 7) / 8;       // 8 waves per block

    const long long tot64 = (long long)n * H1D;
    const long long totC  = (long long)n * CD;
    const long long ew64  = (long long)e * (H1D / 4);
    const long long ewC   = (long long)e * (CD / 4);
    const int blk64  = (int)((tot64 + 255) / 256);
    const int blkC   = (int)((totC + 255) / 256);
    const int blkE64 = (int)((ew64 + 255) / 256);
    const int blkEC  = (int)((ewC + 255) / 256);

    // ---- Pack weights into WMMA B-fragment layout (f16) ----
    pack_w<F_IN, H1D><<<(F_IN * H1D + 255) / 256, B, 0, stream>>>(W1, wb1);
    pack_w<H1D, H2D><<<(H1D * H2D + 255) / 256, B, 0, stream>>>(W2, wb2);
    pack_w<H2D, CD ><<<(H2D * CD  + 255) / 256, B, 0, stream>>>(W3, wb3);

    // ---- Layer 1: support = x @ W1 ; agg = b1 + scatter ----
    gemm_wmma<F_IN, H1D, false><<<gemm_blocks, B, 0, stream>>>(x, wb1, bufA, n);
    init_bias<H1D><<<blk64, B, 0, stream>>>(bufB, b1, tot64);
    spmm_scatter<H1D><<<blkE64, B, 0, stream>>>(bufA, ev, esrc, edst, bufB, ew64);

    // ---- Layer 2: support = relu(agg1) @ W2 ; agg = b2 + scatter ----
    gemm_wmma<H1D, H2D, true><<<gemm_blocks, B, 0, stream>>>(bufB, wb2, bufA, n);
    init_bias<H2D><<<blk64, B, 0, stream>>>(bufB, b2, tot64);
    spmm_scatter<H2D><<<blkE64, B, 0, stream>>>(bufA, ev, esrc, edst, bufB, ew64);

    // ---- Layer 3: support = relu(agg2) @ W3 ; out = b3 + scatter ----
    gemm_wmma<H2D, CD, true><<<gemm_blocks, B, 0, stream>>>(bufB, wb3, bufC, n);
    init_bias<CD><<<blkC, B, 0, stream>>>(out, b3, totC);
    spmm_scatter<CD><<<blkEC, B, 0, stream>>>(bufC, ev, esrc, edst, out, ewC);
}
